// ManifoldAugmentation_81003083202622
// MI455X (gfx1250) — matrix operand order repaired and checked by reading, not compiled, not run on hardware
//
#include <hip/hip_runtime.h>
#include <hip/hip_bf16.h>
#include <stdint.h>

typedef __attribute__((ext_vector_type(16))) __bf16 v16bf;
typedef __attribute__((ext_vector_type(8)))  __bf16 v8bf;
typedef __attribute__((ext_vector_type(8)))  float  v8f;
typedef __attribute__((ext_vector_type(4)))  unsigned int u32x4;
typedef __attribute__((ext_vector_type(4)))  int i32x4;
typedef __attribute__((ext_vector_type(8)))  int i32x8;

#define N_ROWS 8192
#define DIM    256
#define KNB    5            // N_NEIGHBORS
#define TOPK   6            // k+1 (self included, dropped at selection)
#define WAVES  8            // waves per workgroup, each owns 1/8 of the j-sweep
#define JTILES 512          // 8192/16 column tiles
#define JT_PER_WAVE (JTILES / WAVES)
#define LDS_ROW 528         // 512B row + 16B TDM pad -> bank spread
#define STRIP_BYTES (16 * LDS_ROW)

#if defined(__has_builtin)
#if __has_builtin(__builtin_amdgcn_tensor_load_to_lds) && \
    __has_builtin(__builtin_amdgcn_s_wait_tensorcnt)
#define USE_TDM 1
#endif
#endif
#ifndef USE_TDM
#define USE_TDM 0
#endif

__device__ __forceinline__ unsigned short f32_to_bf16_rne(float f) {
  unsigned u = __float_as_uint(f);
  u += 0x7FFFu + ((u >> 16) & 1u);
  return (unsigned short)(u >> 16);
}
__device__ __forceinline__ float bf16u_to_f32(unsigned short h) {
  return __uint_as_float(((unsigned)h) << 16);
}
__device__ __forceinline__ unsigned hash_u32(unsigned v) {
  v ^= v >> 16; v *= 0x7feb352du;
  v ^= v >> 15; v *= 0x846ca68bu;
  v ^= v >> 16;
  return v;
}
// LDS aperture keeps the byte offset in the low 32 bits of a flat address.
__device__ __forceinline__ unsigned lds_off_u32(const void* p) {
  return (unsigned)(uintptr_t)p;
}

#if USE_TDM
// TDM: load one 16-row x 256-col 2-byte tile (8KB) into LDS with 16B/row pad.
// D# built per CDNA5 ISA.
__device__ __forceinline__ void tdm_issue_strip(const unsigned short* gbase,
                                                int j0, unsigned lds_off) {
  unsigned long long ga = (unsigned long long)(uintptr_t)(gbase + (size_t)j0 * DIM);
  u32x4 g0;
  g0[0] = 1u;                                   // count=1, user descriptor
  g0[1] = lds_off;                              // lds_addr (bytes)
  g0[2] = (unsigned)ga;                         // global_addr[31:0]
  g0[3] = (unsigned)(ga >> 32) | (2u << 30);    // global_addr[56:32] | type=2
  i32x8 g1;
  // data_size=1 (2B) | pad_enable | pad_interval=6 (128 DW=512B) | pad_amount=3 (4 DW=16B)
  g1[0] = (int)((1u << 16) | (1u << 20) | (6u << 22) | (3u << 25));
  g1[1] = (int)(256u << 16);    // tensor_dim0 = 256 (bits 63:48)
  g1[2] = (int)(8192u << 16);   // tensor_dim0_hi=0 | tensor_dim1 = 8192 (bits 95:80)
  g1[3] = (int)(256u << 16);    // tensor_dim1_hi=0 | tile_dim0 = 256 (bits 127:112)
  g1[4] = 16;                   // tile_dim1 = 16, tile_dim2 = 0
  g1[5] = 256;                  // tensor_dim0_stride = 256 (low 32)
  g1[6] = 0;
  g1[7] = 0;
  i32x4 z4 = {0, 0, 0, 0};
#if __clang_major__ >= 23
  i32x8 z8 = {0, 0, 0, 0, 0, 0, 0, 0};
  __builtin_amdgcn_tensor_load_to_lds(g0, g1, z4, z4, z8, 0);
#else
  __builtin_amdgcn_tensor_load_to_lds(g0, g1, z4, z4, 0);
#endif
}
#endif

// ---------------------------------------------------------------------------
// Kernel 1: split x into bf16 hi/lo, compute row squared norms.
// ---------------------------------------------------------------------------
__global__ void __launch_bounds__(DIM)
ManifoldAug_prep(const float* __restrict__ x,
                 unsigned short* __restrict__ xhi,
                 unsigned short* __restrict__ xlo,
                 float* __restrict__ sq) {
  __shared__ float red[DIM];
  const int row = blockIdx.x;
  const int t   = threadIdx.x;
  const float v = x[(size_t)row * DIM + t];
  const unsigned short h = f32_to_bf16_rne(v);
  const unsigned short l = f32_to_bf16_rne(v - bf16u_to_f32(h));
  xhi[(size_t)row * DIM + t] = h;
  xlo[(size_t)row * DIM + t] = l;
  red[t] = v * v;
  __syncthreads();
  for (int s = DIM / 2; s > 0; s >>= 1) {
    if (t < s) red[t] += red[t + s];
    __syncthreads();
  }
  if (t == 0) sq[row] = red[0];
}

// ---------------------------------------------------------------------------
// Kernel 2: workgroup = 8 waves sharing one 16-row strip; each wave sweeps
// 64 column tiles with TDM-staged B strips and 3-term split-bf16 WMMA,
// keeps a private top-6, then the lists are merged and the mixup row written.
// ---------------------------------------------------------------------------
__global__ void __launch_bounds__(32 * WAVES)
ManifoldAug_knn(const float* __restrict__ x,
                const unsigned short* __restrict__ xhi,
                const unsigned short* __restrict__ xlo,
                const float* __restrict__ sq,
                float* __restrict__ out) {
#if USE_TDM
  __shared__ __align__(128) unsigned char ldsB[WAVES][2][2][STRIP_BYTES]; // [buf][hi/lo]
#else
  __shared__ __align__(128) unsigned char ldsB[WAVES][1][2][STRIP_BYTES];
#endif
  __shared__ float tileC[WAVES][256];
  __shared__ float mD[WAVES][16][TOPK];
  __shared__ int   mI[WAVES][16][TOPK];
  __shared__ int   selS[16];
  __shared__ float alpS[16];

  const int tid  = threadIdx.x;
  const int lane = tid & 31;
  const int wave = tid >> 5;
  const int ml   = lane & 15;       // M (A) / N (B,C)
  const int kh   = lane >> 4;       // K-half select
  const int rowbase = blockIdx.x * 16;

  // ---- A fragments (all waves need the same 16 rows), held in VGPRs.
  v16bf ahi[8], alo[8];
  {
    const unsigned short* ph = xhi + (size_t)(rowbase + ml) * DIM;
    const unsigned short* pl = xlo + (size_t)(rowbase + ml) * DIM;
#pragma unroll
    for (int kt = 0; kt < 8; ++kt) {
      const int k0 = kt * 32;
      v8bf h0 = *(const v8bf*)(ph + k0 + 16 * kh);
      v8bf h1 = *(const v8bf*)(ph + k0 + 16 * kh + 8);
      v8bf l0 = *(const v8bf*)(pl + k0 + 16 * kh);
      v8bf l1 = *(const v8bf*)(pl + k0 + 16 * kh + 8);
#pragma unroll
      for (int e = 0; e < 8; ++e) {
        ahi[kt][e] = h0[e]; ahi[kt][8 + e] = h1[e];
        alo[kt][e] = l0[e]; alo[kt][8 + e] = l1[e];
      }
    }
  }

  float bd[TOPK];
  int   bi[TOPK];
#pragma unroll
  for (int s = 0; s < TOPK; ++s) { bd[s] = 3.4e38f; bi[s] = -1; }

  const int t0 = wave * JT_PER_WAVE;

#if USE_TDM
  const unsigned offH[2] = { lds_off_u32(&ldsB[wave][0][0][0]),
                             lds_off_u32(&ldsB[wave][1][0][0]) };
  const unsigned offL[2] = { lds_off_u32(&ldsB[wave][0][1][0]),
                             lds_off_u32(&ldsB[wave][1][1][0]) };
  tdm_issue_strip(xhi, t0 * 16, offH[0]);
  tdm_issue_strip(xlo, t0 * 16, offL[0]);
#endif

  for (int tt = 0; tt < JT_PER_WAVE; ++tt) {
    const int j0 = (t0 + tt) * 16;
#if USE_TDM
    const int buf = tt & 1;
    if (tt + 1 < JT_PER_WAVE) {
      // prior ds reads of the buffer we are about to overwrite must retire
      asm volatile("s_wait_dscnt 0x0" ::: "memory");
      tdm_issue_strip(xhi, (t0 + tt + 1) * 16, offH[(tt + 1) & 1]);
      tdm_issue_strip(xlo, (t0 + tt + 1) * 16, offL[(tt + 1) & 1]);
      __builtin_amdgcn_s_wait_tensorcnt((short)2);  // current pair landed
    } else {
      __builtin_amdgcn_s_wait_tensorcnt((short)0);
    }
#else
    const int buf = 0;
    // cooperative wave-local copy, same padded layout as the TDM path
    unsigned char* dh = &ldsB[wave][0][0][0];
    unsigned char* dl = &ldsB[wave][0][1][0];
    for (int c = lane; c < 512; c += 32) {     // 16 rows x 32 x 16B chunks
      const int r = c >> 5, kc = c & 31;
      *(u32x4*)(dh + r * LDS_ROW + kc * 16) =
          *(const u32x4*)(xhi + (size_t)(j0 + r) * DIM + kc * 8);
      *(u32x4*)(dl + r * LDS_ROW + kc * 16) =
          *(const u32x4*)(xlo + (size_t)(j0 + r) * DIM + kc * 8);
    }
    // LDS ops from one wave execute in order: stores land before later loads.
#endif

    // B fragments from LDS: lane -> col ml, K span 16*kh + kt*32 .. +15
    const unsigned char* bhb = &ldsB[wave][buf][0][0];
    const unsigned char* blb = &ldsB[wave][buf][1][0];
    const int boff = ml * LDS_ROW + 32 * kh;   // bytes
    v8f acc = {};
#pragma unroll
    for (int kt = 0; kt < 8; ++kt) {
      v16bf bh = *(const v16bf*)(bhb + boff + kt * 64);
      v16bf bl = *(const v16bf*)(blb + boff + kt * 64);
      acc = __builtin_amdgcn_wmma_f32_16x16x32_bf16(
          false, ahi[kt], false, bh, (short)0, acc, false, false);
      acc = __builtin_amdgcn_wmma_f32_16x16x32_bf16(
          false, ahi[kt], false, bl, (short)0, acc, false, false);
      acc = __builtin_amdgcn_wmma_f32_16x16x32_bf16(
          false, alo[kt], false, bh, (short)0, acc, false, false);
    }

    // rank value: sq[j] - 2*dot  (sq[i] constant per row -> same ordering)
    const float sj = sq[j0 + ml];
#pragma unroll
    for (int v = 0; v < 8; ++v) {
      const int m = v + 8 * kh;                // C layout: M = v + 8*(lane>>4)
      tileC[wave][m * 16 + ml] = sj - 2.0f * acc[v];
    }
    // wave-local LDS in-order: scan sees the stores above
    if (lane < 16) {
#pragma unroll
      for (int n = 0; n < 16; ++n) {
        const float d  = tileC[wave][lane * 16 + n];
        const int  idx = j0 + n;
        if (d < bd[TOPK - 1]) {
          bd[TOPK - 1] = d; bi[TOPK - 1] = idx;
#pragma unroll
          for (int s = TOPK - 1; s > 0; --s) {
            if (bd[s] < bd[s - 1]) {
              float td = bd[s]; bd[s] = bd[s - 1]; bd[s - 1] = td;
              int   ti = bi[s]; bi[s] = bi[s - 1]; bi[s - 1] = ti;
            }
          }
        }
      }
    }
  }

  // ---- publish per-wave partial top-6 lists
  if (lane < 16) {
#pragma unroll
    for (int s = 0; s < TOPK; ++s) { mD[wave][lane][s] = bd[s]; mI[wave][lane][s] = bi[s]; }
  }
  __syncthreads();

  // ---- wave 0 merges 8 partial lists per row, selects neighbor + alpha
  if (wave == 0 && lane < 16) {
    const int row = rowbase + lane;
    float fd[TOPK]; int fi[TOPK];
#pragma unroll
    for (int s = 0; s < TOPK; ++s) { fd[s] = 3.4e38f; fi[s] = -1; }
    for (int w = 0; w < WAVES; ++w) {
#pragma unroll
      for (int s = 0; s < TOPK; ++s) {
        const float d = mD[w][lane][s];
        const int idx = mI[w][lane][s];
        if (d < fd[TOPK - 1]) {
          fd[TOPK - 1] = d; fi[TOPK - 1] = idx;
#pragma unroll
          for (int q = TOPK - 1; q > 0; --q) {
            if (fd[q] < fd[q - 1]) {
              float td = fd[q]; fd[q] = fd[q - 1]; fd[q - 1] = td;
              int   ti = fi[q]; fi[q] = fi[q - 1]; fi[q - 1] = ti;
            }
          }
        }
      }
    }
    int nbr[KNB];
#pragma unroll
    for (int s = 0; s < KNB; ++s) nbr[s] = fi[1];
    int c = 0;
#pragma unroll
    for (int s = 0; s < TOPK; ++s) {
      if (fi[s] != row && c < KNB) { nbr[c] = fi[s]; ++c; }
    }
    const unsigned h1 = hash_u32((unsigned)(row * 2 + 1));
    const unsigned h2 = hash_u32((unsigned)(row * 2 + 2) ^ 0x9e3779b9u);
    selS[lane] = nbr[h1 % (unsigned)KNB];
    alpS[lane] = (float)(h2 >> 8) * (1.0f / 16777216.0f);
  }
  __syncthreads();

  // ---- write augmented rows with all 256 threads
  for (int idx = tid; idx < 16 * DIM; idx += 32 * WAVES) {
    const int m = idx >> 8;
    const int t = idx & (DIM - 1);
    const int row = rowbase + m;
    const float al = alpS[m];
    out[(size_t)(N_ROWS + row) * DIM + t] =
        (1.0f - al) * x[(size_t)row * DIM + t] + al * x[(size_t)selS[m] * DIM + t];
  }
}

// ---------------------------------------------------------------------------
extern "C" void kernel_launch(void* const* d_in, const int* in_sizes, int n_in,
                              void* d_out, int out_size, void* d_ws, size_t ws_size,
                              hipStream_t stream) {
  (void)in_sizes; (void)n_in; (void)out_size; (void)ws_size;
  const float* x = (const float*)d_in[0];
  float* out = (float*)d_out;

  // workspace: xhi (4MB) | xlo (4MB) | sq (32KB)
  unsigned short* xhi = (unsigned short*)d_ws;
  unsigned short* xlo = xhi + (size_t)N_ROWS * DIM;
  float*          sqn = (float*)(xlo + (size_t)N_ROWS * DIM);

  ManifoldAug_prep<<<N_ROWS, DIM, 0, stream>>>(x, xhi, xlo, sqn);

  hipMemcpyAsync(out, x, (size_t)N_ROWS * DIM * sizeof(float),
                 hipMemcpyDeviceToDevice, stream);

  ManifoldAug_knn<<<N_ROWS / 16, 32 * WAVES, 0, stream>>>(x, xhi, xlo, sqn, out);
}
